// GNNLayer_7215545057969
// MI455X (gfx1250) — matrix operand order repaired and checked by reading, not compiled, not run on hardware
//
#include <hip/hip_runtime.h>
#include <cstdint>

#define NND 32768
#define NED 524288
#define HID 128

typedef __attribute__((ext_vector_type(16))) __bf16 v16bf;
typedef __attribute__((ext_vector_type(8)))  float  v8f;

__device__ __forceinline__ v8f wmma_bf16(v16bf a, v16bf b, v8f c) {
  // (neg_a, A, neg_b, B, c_mod, C, reuse_a, reuse_b)
  return __builtin_amdgcn_wmma_f32_16x16x32_bf16(false, a, false, b, (short)0, c, false, false);
}

__device__ __forceinline__ float silu_f(float v) {
  return v / (1.0f + __expf(-v));
}

// ---- Fragment loaders (layouts per CDNA5 ISA 7.12.2) -------------
// 16-bit A 16x32: lanes 0-15 row M=lane, lanes 16-31 row M=lane-16.
//   element e -> K = e + 8*half + (e>=8 ? 8 : 0)
__device__ __forceinline__ v16bf load_a_f32(const float* __restrict__ row, int half) {
  v16bf a;
  const float* p0 = row + 8 * half;        // e=0..7  -> K = 8*half + e
  const float* p1 = row + 16 + 8 * half;   // e=8..15 -> K = 16 + 8*half + (e-8)
#pragma unroll
  for (int i = 0; i < 8; ++i) a[i] = (__bf16)p0[i];
#pragma unroll
  for (int i = 0; i < 8; ++i) a[8 + i] = (__bf16)p1[i];
  return a;
}

__device__ __forceinline__ v16bf load_a_bf16(const __bf16* row, int half) {
  v16bf a;
  const __bf16* p0 = row + 8 * half;
  const __bf16* p1 = row + 16 + 8 * half;
#pragma unroll
  for (int i = 0; i < 8; ++i) a[i] = p0[i];
#pragma unroll
  for (int i = 0; i < 8; ++i) a[8 + i] = p1[i];
  return a;
}

// 16-bit B 32x16: column N = lane%16; element e -> K = e + 8*half + (e>=8 ? 8 : 0).
// `col` points at wt[n]*K + k_window (transposed, K-contiguous weights).
__device__ __forceinline__ v16bf load_b(const __bf16* __restrict__ col, int half) {
  v16bf b;
  const __bf16* p0 = col + 8 * half;
  const __bf16* p1 = col + 16 + 8 * half;
#pragma unroll
  for (int i = 0; i < 8; ++i) b[i] = p0[i];
#pragma unroll
  for (int i = 0; i < 8; ++i) b[8 + i] = p1[i];
  return b;
}

// ---- Utility kernels --------------------------------------------------------
__global__ void zero_kernel(float* __restrict__ p, int n) {
  int i = blockIdx.x * 256 + threadIdx.x;
  if (i < n) p[i] = 0.0f;
}

// Transpose + convert weights to bf16, out-feature-major (K contiguous).
__global__ void prep_kernel(const float* __restrict__ w1, const float* __restrict__ w2,
                            const float* __restrict__ wu1, const float* __restrict__ wu2,
                            __bf16* __restrict__ w1t, __bf16* __restrict__ w2t,
                            __bf16* __restrict__ wu1t, __bf16* __restrict__ wu2t) {
  int i = blockIdx.x * 256 + threadIdx.x;
  if (i < 384 * 128) { int k = i / 128, n = i % 128; w1t[n * 384 + k]  = (__bf16)w1[i]; }
  if (i < 128 * 128) { int k = i / 128, n = i % 128; w2t[n * 128 + k]  = (__bf16)w2[i]; }
  if (i < 128 * 256) { int k = i / 256, n = i % 256; wu1t[n * 128 + k] = (__bf16)wu1[i]; }
  if (i < 256 * 128) { int k = i / 128, n = i % 128; wu2t[n * 256 + k] = (__bf16)wu2[i]; }
}

// Convert node features to bf16 once (reused ~32x by edge gathers, L2-resident).
__global__ void prep_x_kernel(const float* __restrict__ x, __bf16* __restrict__ x_bf, int n) {
  int i = blockIdx.x * 256 + threadIdx.x;
  if (i < n) x_bf[i] = (__bf16)x[i];
}

// ---- Edge message kernel: 1 wave per 32-edge tile (M=32, two A tiles) -------
__global__ __launch_bounds__(128) void edge_kernel(
    const __bf16* __restrict__ x_bf, const float* __restrict__ eattr,
    const int* __restrict__ send, const int* __restrict__ recv,
    const __bf16* __restrict__ w1t, const float* __restrict__ b1,
    const __bf16* __restrict__ w2t, const float* __restrict__ b2,
    float* __restrict__ msg_out, float* __restrict__ sums, float* __restrict__ counts) {
  __shared__ __bf16 h_lds[4][32][HID];

  const int lane = threadIdx.x & 31;
  const int wave = threadIdx.x >> 5;
  const int half = lane >> 4;
  const int lm   = lane & 15;
  const int tile = blockIdx.x * 4 + wave;
  const int e0   = tile * 32;

  // scatter-mean denominators (one lane per edge, 32 edges)
  {
    int rv = recv[e0 + lane];
    __hip_atomic_fetch_add(&counts[rv], 1.0f, __ATOMIC_RELAXED, __HIP_MEMORY_SCOPE_AGENT);
  }

  // ---- layer 1: [32,384] @ [384,128], K in 12 steps of 32 ------------------
  v8f acc[2][8];
#pragma unroll
  for (int t = 0; t < 2; ++t)
#pragma unroll
    for (int nt = 0; nt < 8; ++nt)
#pragma unroll
      for (int r = 0; r < 8; ++r) acc[t][nt][r] = 0.0f;

  const __bf16* srcS[2];     // x_bf[send], cols [0,128)
  const __bf16* srcR[2];     // x_bf[recv], cols [128,256)
  const float*  srcE[2];     // edge_attr,  cols [256,384)
#pragma unroll
  for (int t = 0; t < 2; ++t) {
    int me = e0 + t * 16 + lm;                       // this lane's A row (edge)
    srcS[t] = x_bf + (size_t)send[me] * HID;
    srcR[t] = x_bf + (size_t)recv[me] * HID;
    srcE[t] = eattr + (size_t)me * HID;
  }

  // K steps 0..7: bf16 gathered node features (no conversion needed)
#pragma unroll
  for (int kk = 0; kk < 8; ++kk) {
    v16bf a[2];
#pragma unroll
    for (int t = 0; t < 2; ++t) {
      const __bf16* base = (kk < 4) ? srcS[t] : srcR[t];
      a[t] = load_a_bf16(base + (kk & 3) * 32, half);
    }
#pragma unroll
    for (int nt = 0; nt < 8; ++nt) {
      v16bf b = load_b(w1t + (size_t)(nt * 16 + lm) * 384 + kk * 32, half);
      acc[0][nt] = wmma_bf16(a[0], b, acc[0][nt]);
      acc[1][nt] = wmma_bf16(a[1], b, acc[1][nt]);
    }
  }
  // K steps 8..11: f32 edge attributes (single-use, convert in flight)
#pragma unroll
  for (int kk = 8; kk < 12; ++kk) {
    v16bf a[2];
#pragma unroll
    for (int t = 0; t < 2; ++t)
      a[t] = load_a_f32(srcE[t] + (kk - 8) * 32, half);
#pragma unroll
    for (int nt = 0; nt < 8; ++nt) {
      v16bf b = load_b(w1t + (size_t)(nt * 16 + lm) * 384 + kk * 32, half);
      acc[0][nt] = wmma_bf16(a[0], b, acc[0][nt]);
      acc[1][nt] = wmma_bf16(a[1], b, acc[1][nt]);
    }
  }

  // bias + silu -> bf16 h in LDS (D layout: M = r + 8*half, N = nt*16 + lm)
#pragma unroll
  for (int nt = 0; nt < 8; ++nt) {
    float bv = b1[nt * 16 + lm];
#pragma unroll
    for (int t = 0; t < 2; ++t)
#pragma unroll
      for (int r = 0; r < 8; ++r)
        h_lds[wave][t * 16 + r + 8 * half][nt * 16 + lm] = (__bf16)silu_f(acc[t][nt][r] + bv);
  }
  __syncthreads();

  // ---- layer 2: [32,128] @ [128,128] ---------------------------------------
  v8f acc2[2][8];
#pragma unroll
  for (int t = 0; t < 2; ++t)
#pragma unroll
    for (int nt = 0; nt < 8; ++nt)
#pragma unroll
      for (int r = 0; r < 8; ++r) acc2[t][nt][r] = 0.0f;

#pragma unroll
  for (int kk = 0; kk < 4; ++kk) {
    v16bf a0 = load_a_bf16(&h_lds[wave][lm][kk * 32], half);
    v16bf a1 = load_a_bf16(&h_lds[wave][16 + lm][kk * 32], half);
#pragma unroll
    for (int nt = 0; nt < 8; ++nt) {
      v16bf b = load_b(w2t + (size_t)(nt * 16 + lm) * 128 + kk * 32, half);
      acc2[0][nt] = wmma_bf16(a0, b, acc2[0][nt]);
      acc2[1][nt] = wmma_bf16(a1, b, acc2[1][nt]);
    }
  }

  // bias + silu -> msg out + scatter-sum into receiving nodes
  int rv[2][8];
#pragma unroll
  for (int t = 0; t < 2; ++t)
#pragma unroll
    for (int r = 0; r < 8; ++r) rv[t][r] = recv[e0 + t * 16 + r + 8 * half];

#pragma unroll
  for (int nt = 0; nt < 8; ++nt) {
    int n = nt * 16 + lm;
    float bv = b2[n];
#pragma unroll
    for (int t = 0; t < 2; ++t)
#pragma unroll
      for (int r = 0; r < 8; ++r) {
        float v = silu_f(acc2[t][nt][r] + bv);
        msg_out[(size_t)(e0 + t * 16 + r + 8 * half) * HID + n] = v;
        __hip_atomic_fetch_add(&sums[(size_t)rv[t][r] * HID + n], v,
                               __ATOMIC_RELAXED, __HIP_MEMORY_SCOPE_AGENT);
      }
  }
}

// ---- Node update kernel: 1 wave per 16-node tile ----------------------------
__global__ __launch_bounds__(64) void node_kernel(
    const float* __restrict__ x, const float* __restrict__ sums,
    const float* __restrict__ counts,
    const __bf16* __restrict__ wu1t, const float* __restrict__ bu1,
    const __bf16* __restrict__ wu2t, const float* __restrict__ bu2,
    float* __restrict__ xo_out) {
  __shared__ float  xo_f[2][16][HID];
  __shared__ __bf16 xo_h[2][16][HID];
  __shared__ __bf16 t_h[2][16][2 * HID];

  const int lane = threadIdx.x & 31;
  const int wave = threadIdx.x >> 5;
  const int half = lane >> 4;
  const int lm   = lane & 15;
  const int tile = blockIdx.x * 2 + wave;
  const int n0   = tile * 16;

  // xo = x + sums / max(counts, 1)
#pragma unroll 4
  for (int j = 0; j < 64; ++j) {
    int idx = lane + 32 * j;            // 0..2047
    int row = idx >> 7, col = idx & 127;
    float c = counts[n0 + row];
    size_t off = (size_t)(n0 + row) * HID + col;
    float v = x[off] + sums[off] / fmaxf(c, 1.0f);
    xo_f[wave][row][col] = v;
    xo_h[wave][row][col] = (__bf16)v;
  }
  __syncthreads();

  // ---- update layer 1: [16,128] @ [128,256] --------------------------------
#pragma unroll
  for (int nt = 0; nt < 16; ++nt) {
    v8f acc;
#pragma unroll
    for (int r = 0; r < 8; ++r) acc[r] = 0.0f;
#pragma unroll
    for (int kk = 0; kk < 4; ++kk) {
      v16bf a = load_a_bf16(&xo_h[wave][lm][kk * 32], half);
      v16bf b = load_b(wu1t + (size_t)(nt * 16 + lm) * 128 + kk * 32, half);
      acc = wmma_bf16(a, b, acc);
    }
    float bv = bu1[nt * 16 + lm];
#pragma unroll
    for (int r = 0; r < 8; ++r)
      t_h[wave][r + 8 * half][nt * 16 + lm] = (__bf16)silu_f(acc[r] + bv);
  }
  __syncthreads();

  // ---- update layer 2: [16,256] @ [256,128] + residual ---------------------
#pragma unroll
  for (int nt = 0; nt < 8; ++nt) {
    v8f acc;
#pragma unroll
    for (int r = 0; r < 8; ++r) acc[r] = 0.0f;
#pragma unroll
    for (int kk = 0; kk < 8; ++kk) {
      v16bf a = load_a_bf16(&t_h[wave][lm][kk * 32], half);
      v16bf b = load_b(wu2t + (size_t)(nt * 16 + lm) * 256 + kk * 32, half);
      acc = wmma_bf16(a, b, acc);
    }
    int n = nt * 16 + lm;
    float bv = bu2[n];
#pragma unroll
    for (int r = 0; r < 8; ++r) {
      int M = r + 8 * half;
      xo_out[(size_t)(n0 + M) * HID + n] = acc[r] + bv + xo_f[wave][M][n];
    }
  }
}

extern "C" void kernel_launch(void* const* d_in, const int* in_sizes, int n_in,
                              void* d_out, int out_size, void* d_ws, size_t ws_size,
                              hipStream_t stream) {
  (void)in_sizes; (void)n_in; (void)out_size; (void)ws_size;
  const float* x     = (const float*)d_in[0];
  const float* eattr = (const float*)d_in[1];
  const int*   edges = (const int*)d_in[2];
  const float* w1  = (const float*)d_in[3];
  const float* b1  = (const float*)d_in[4];
  const float* w2  = (const float*)d_in[5];
  const float* b2  = (const float*)d_in[6];
  const float* wu1 = (const float*)d_in[7];
  const float* bu1 = (const float*)d_in[8];
  const float* wu2 = (const float*)d_in[9];
  const float* bu2 = (const float*)d_in[10];

  const int* send = edges;
  const int* recv = edges + NED;

  float* out_xo  = (float*)d_out;                       // [NND, HID]
  float* out_msg = out_xo + (size_t)NND * HID;          // [NED, HID]

  // workspace carve-up
  float*  sums   = (float*)d_ws;                        // NND*HID f32
  float*  counts = sums + (size_t)NND * HID;            // NND f32
  __bf16* w1t    = (__bf16*)(counts + NND);             // [128][384]
  __bf16* w2t    = w1t + 384 * 128;                     // [128][128]
  __bf16* wu1t   = w2t + 128 * 128;                     // [256][128]
  __bf16* wu2t   = wu1t + 128 * 256;                    // [128][256]
  __bf16* x_bf   = wu2t + 256 * 128;                    // [NND][HID] bf16

  const int nz = NND * HID + NND;
  zero_kernel<<<(nz + 255) / 256, 256, 0, stream>>>(sums, nz);
  prep_kernel<<<(384 * 128 + 255) / 256, 256, 0, stream>>>(
      w1, w2, wu1, wu2, w1t, w2t, wu1t, wu2t);
  prep_x_kernel<<<(NND * HID + 255) / 256, 256, 0, stream>>>(x, x_bf, NND * HID);

  // E/32 = 16384 edge tiles, 4 waves per 128-thread block
  edge_kernel<<<NED / 32 / 4, 128, 0, stream>>>(
      x_bf, eattr, send, recv, w1t, b1, w2t, b2, out_msg, sums, counts);

  // N/16 = 2048 node tiles, 2 waves per 64-thread block
  node_kernel<<<NND / 16 / 2, 64, 0, stream>>>(
      x, sums, counts, wu1t, bu1, wu2t, bu2, out_xo);
}
